// DropoutAfterBLSTM_23725399343713
// MI455X (gfx1250) — matrix operand order repaired and checked by reading, not compiled, not run on hardware
//
#include <hip/hip_runtime.h>
#include <hip/hip_bf16.h>
#include <math.h>

// ---------------------------------------------------------------------------
// BiLSTM (eval mode), B=64 S=512 E=1024 H=1024, fp32 in/out.
// Strategy (MI455X / gfx1250, wave32, WMMA):
//   * pack [Wih;Whh] -> W_cat[dir][N=4096][K=2048] bf16 (transposed so B-frag
//     loads are 32B contiguous per lane), pack x -> bf16.
//   * persistent scan kernel: 2 dirs x 32 WGs x 256 thr. Each WG owns
//     M=64(batch) x N=32 h-cols; each wave one 16x16 h-tile + 4 gate accs.
//   * per step: gates = [x_s, h_{t-1}] @ W_cat (K=2048) via
//     v_wmma_f32_16x16x32_bf16; x-part computed BEFORE the grid barrier to
//     hide sync latency. c kept in registers; h double-buffered in global
//     (L2-resident, 128KB/dir). Release/acquire atomic counter per step.
//   * gate activations use gfx1250 TRANS ops (v_exp_f32 / v_rcp_f32 /
//     v_tanh_f32) to shorten the serial per-step critical path.
// ---------------------------------------------------------------------------

typedef __bf16 bf16;
typedef bf16  v16bf  __attribute__((ext_vector_type(16)));
typedef bf16  bf16x8 __attribute__((ext_vector_type(8)));
typedef float v8f    __attribute__((ext_vector_type(8)));

#define B_   64
#define S_   512
#define E_   1024
#define H_   1024
#define G4_  4096          // 4*H
#define K2_  2048          // E + H
#define NBLK 32            // workgroups per direction

// workspace layout (bytes)
#define OFF_WCAT  ((size_t)0)                                  // 2*4096*2048 bf16 = 32MB
#define OFF_BCAT  (OFF_WCAT + (size_t)2*G4_*K2_*2)             // 2*4096 f32
#define OFF_XB    (OFF_BCAT + (size_t)2*G4_*4)                 // 64*512*1024 bf16 = 64MB
#define OFF_HBUF  (OFF_XB   + (size_t)B_*S_*E_*2)              // 2par*2dir*64*1024 bf16
#define OFF_CNT   (OFF_HBUF + (size_t)2*2*B_*H_*2)             // 2*512 int

// ---- fast transcendentals (gfx1250 TRANS unit) ----------------------------
#if __has_builtin(__builtin_amdgcn_tanhf)
__device__ __forceinline__ float fast_tanh(float x) { return __builtin_amdgcn_tanhf(x); }
#elif __has_builtin(__builtin_amdgcn_tanh_f32)
__device__ __forceinline__ float fast_tanh(float x) { return __builtin_amdgcn_tanh_f32(x); }
#else
__device__ __forceinline__ float fast_tanh(float x) { return tanhf(x); }
#endif

#if __has_builtin(__builtin_amdgcn_rcpf)
__device__ __forceinline__ float fast_rcp(float x) { return __builtin_amdgcn_rcpf(x); }
#else
__device__ __forceinline__ float fast_rcp(float x) { return 1.0f / x; }
#endif

__device__ __forceinline__ float sigf(float x) { return fast_rcp(1.0f + __expf(-x)); }

// A fragment (16x32 bf16): lane half=l/16 selects K subrange; two 16B loads.
__device__ __forceinline__ v16bf load_a(const bf16* p) {
  bf16x8 lo = *reinterpret_cast<const bf16x8*>(p);
  bf16x8 hi = *reinterpret_cast<const bf16x8*>(p + 8);
  return __builtin_shufflevector(lo, hi, 0,1,2,3,4,5,6,7,8,9,10,11,12,13,14,15);
}
// B fragment (32x16 bf16) from K-contiguous packed weights: one 32B load.
__device__ __forceinline__ v16bf load_b(const bf16* p) {
  return *reinterpret_cast<const v16bf*>(p);
}
__device__ __forceinline__ v8f mma(v16bf a, v16bf b, v8f c) {
  return __builtin_amdgcn_wmma_f32_16x16x32_bf16(false, a, false, b, (short)0, c, false, false);
}

// -------------------------------- pack kernels -----------------------------

__global__ void pack_wcat_kernel(const float* __restrict__ WihF, const float* __restrict__ WhhF,
                                 const float* __restrict__ WihB, const float* __restrict__ WhhB,
                                 bf16* __restrict__ wcat) {
  size_t idx = (size_t)blockIdx.x * blockDim.x + threadIdx.x;
  const size_t total = (size_t)2 * G4_ * K2_;
  if (idx >= total) return;
  int k   = (int)(idx % K2_);
  int n   = (int)((idx / K2_) % G4_);
  int dir = (int)(idx / ((size_t)K2_ * G4_));
  const float* Wih = dir ? WihB : WihF;   // [E,4H]
  const float* Whh = dir ? WhhB : WhhF;   // [H,4H]
  float v = (k < E_) ? Wih[(size_t)k * G4_ + n] : Whh[(size_t)(k - E_) * G4_ + n];
  wcat[idx] = (bf16)v;
}

__global__ void pack_bias_kernel(const float* __restrict__ bihF, const float* __restrict__ bhhF,
                                 const float* __restrict__ bihB, const float* __restrict__ bhhB,
                                 float* __restrict__ bcat) {
  int idx = blockIdx.x * blockDim.x + threadIdx.x;
  if (idx >= 2 * G4_) return;
  int dir = idx / G4_, n = idx % G4_;
  bcat[idx] = dir ? (bihB[n] + bhhB[n]) : (bihF[n] + bhhF[n]);
}

__global__ void pack_x_kernel(const float* __restrict__ x, bf16* __restrict__ xb) {
  size_t idx = (size_t)blockIdx.x * blockDim.x + threadIdx.x;
  if (idx >= (size_t)B_ * S_ * E_) return;
  xb[idx] = (bf16)x[idx];
}

__global__ void init_state_kernel(bf16* __restrict__ hbuf, int* __restrict__ cnt) {
  size_t idx = (size_t)blockIdx.x * blockDim.x + threadIdx.x;
  const size_t nh = (size_t)2 * 2 * B_ * H_;
  if (idx < nh) hbuf[idx] = (bf16)0.0f;
  if (idx < 2 * S_) cnt[idx] = 0;
}

// -------------------------------- scan kernel ------------------------------

__global__ __launch_bounds__(256)
void bilstm_scan_kernel(const bf16* __restrict__ xb,     // [B][S][E] bf16
                        const bf16* __restrict__ wcat,   // [dir][N=4H][K=E+H] bf16
                        const float* __restrict__ bcat,  // [dir][4H]
                        bf16* __restrict__ hbuf,         // [par][dir][B][H] bf16
                        int* cnt,                        // [dir][S]
                        float* __restrict__ out)         // [B][S][2H]
{
  const int dir  = blockIdx.x / NBLK;
  const int cb   = blockIdx.x % NBLK;
  const int wave = threadIdx.x >> 5;
  const int lane = threadIdx.x & 31;
  const int mt   = wave & 3;       // M tile: batch rows 16*mt..+15
  const int nt   = wave >> 2;      // N tile within WG chunk (0..1)
  const int half = lane >> 4;
  const int l15  = lane & 15;

  const int col    = cb * 32 + nt * 16 + l15;   // this thread's h-column
  const int b_row  = mt * 16 + l15;             // batch row this lane loads (A)
  const int m_base = mt * 16 + half * 8;        // acc rows: m_base + r

  // B fragment row pointers for the 4 gates (i,f,g,o at N offsets 0,H,2H,3H)
  const bf16* Wd  = wcat + (size_t)dir * G4_ * K2_;
  const bf16* bp0 = Wd + ((size_t)(0 * H_ + col)) * K2_ + half * 16;
  const bf16* bp1 = Wd + ((size_t)(1 * H_ + col)) * K2_ + half * 16;
  const bf16* bp2 = Wd + ((size_t)(2 * H_ + col)) * K2_ + half * 16;
  const bf16* bp3 = Wd + ((size_t)(3 * H_ + col)) * K2_ + half * 16;

  const float* bc = bcat + dir * G4_;
  const float bi = bc[0 * H_ + col];
  const float bf = bc[1 * H_ + col];
  const float bg = bc[2 * H_ + col];
  const float bo = bc[3 * H_ + col];

  float cst[8];
  #pragma unroll
  for (int r = 0; r < 8; ++r) cst[r] = 0.0f;

  int* cbase = cnt + dir * S_;

  for (int t = 0; t < S_; ++t) {
    const int s_x = dir ? (S_ - 1 - t) : t;   // x index == output time index

    v8f a0 = {}, a1 = {}, a2 = {}, a3 = {};

    // ---- phase X: x_s @ Wih  (independent of h -> computed before barrier)
    const bf16* xrow = xb + (size_t)b_row * (S_ * E_) + (size_t)s_x * E_ + half * 16;
    #pragma unroll 4
    for (int kk = 0; kk < E_; kk += 32) {
      v16bf av = load_a(xrow + kk);
      a0 = mma(av, load_b(bp0 + kk), a0);
      a1 = mma(av, load_b(bp1 + kk), a1);
      a2 = mma(av, load_b(bp2 + kk), a2);
      a3 = mma(av, load_b(bp3 + kk), a3);
    }

    // prefetch next step's x rows while we wait (gfx1250 global_prefetch_b8)
    if (t + 1 < S_) {
      const int s_nx = dir ? (S_ - 2 - t) : (t + 1);
      __builtin_prefetch((const void*)(xb + (size_t)b_row * (S_ * E_) + (size_t)s_nx * E_), 0, 0);
    }

    // ---- wait for h_{t-1} from all WGs of this direction
    if (t > 0) {
      int* cw = cbase + (t - 1);
      while (__hip_atomic_load(cw, __ATOMIC_ACQUIRE, __HIP_MEMORY_SCOPE_AGENT) < NBLK)
        __builtin_amdgcn_s_sleep(0);
    }

    // ---- phase H: h_{t-1} @ Whh  (Whh occupies K rows E_..E_+H_-1 of W_cat)
    const bf16* hrow = hbuf + ((((size_t)(t & 1)) * 2 + dir) * B_ + b_row) * H_ + half * 16;
    #pragma unroll 4
    for (int kk = 0; kk < H_; kk += 32) {
      v16bf av = load_a(hrow + kk);
      a0 = mma(av, load_b(bp0 + E_ + kk), a0);
      a1 = mma(av, load_b(bp1 + E_ + kk), a1);
      a2 = mma(av, load_b(bp2 + E_ + kk), a2);
      a3 = mma(av, load_b(bp3 + E_ + kk), a3);
    }

    // ---- gate math; c lives in registers, h written to dbl-buffer + output
    bf16* hout = hbuf + (((size_t)((t + 1) & 1)) * 2 + dir) * ((size_t)B_ * H_);
    #pragma unroll
    for (int r = 0; r < 8; ++r) {
      const int b = m_base + r;
      float gi = a0[r] + bi;
      float gf = a1[r] + bf;
      float gg = a2[r] + bg;
      float go = a3[r] + bo;
      float cc = sigf(gf) * cst[r] + sigf(gi) * fast_tanh(gg);
      cst[r] = cc;
      float hh = sigf(go) * fast_tanh(cc);
      out[((size_t)b * S_ + s_x) * (2 * H_) + dir * H_ + col] = hh;
      hout[(size_t)b * H_ + col] = (bf16)hh;
    }

    // ---- signal h_t complete (release), ordered after stores
    __threadfence();
    __syncthreads();
    if (threadIdx.x == 0)
      __hip_atomic_fetch_add(cbase + t, 1, __ATOMIC_RELEASE, __HIP_MEMORY_SCOPE_AGENT);
  }
}

// -------------------------------- launcher ---------------------------------

extern "C" void kernel_launch(void* const* d_in, const int* in_sizes, int n_in,
                              void* d_out, int out_size, void* d_ws, size_t ws_size,
                              hipStream_t stream) {
  const float* x     = (const float*)d_in[0];
  const float* WihF  = (const float*)d_in[1];
  const float* bihF  = (const float*)d_in[2];
  const float* WhhF  = (const float*)d_in[3];
  const float* bhhF  = (const float*)d_in[4];
  const float* WihB  = (const float*)d_in[5];
  const float* bihB  = (const float*)d_in[6];
  const float* WhhB  = (const float*)d_in[7];
  const float* bhhB  = (const float*)d_in[8];
  float* out = (float*)d_out;

  char* ws = (char*)d_ws;
  bf16*  wcat = (bf16*)(ws + OFF_WCAT);
  float* bcat = (float*)(ws + OFF_BCAT);
  bf16*  xb   = (bf16*)(ws + OFF_XB);
  bf16*  hbuf = (bf16*)(ws + OFF_HBUF);
  int*   cnt  = (int*)(ws + OFF_CNT);

  {
    size_t total = (size_t)2 * G4_ * K2_;
    pack_wcat_kernel<<<(unsigned)((total + 255) / 256), 256, 0, stream>>>(WihF, WhhF, WihB, WhhB, wcat);
  }
  pack_bias_kernel<<<(2 * G4_ + 255) / 256, 256, 0, stream>>>(bihF, bhhF, bihB, bhhB, bcat);
  {
    size_t total = (size_t)B_ * S_ * E_;
    pack_x_kernel<<<(unsigned)((total + 255) / 256), 256, 0, stream>>>(x, xb);
  }
  {
    size_t total = (size_t)2 * 2 * B_ * H_;
    init_state_kernel<<<(unsigned)((total + 255) / 256), 256, 0, stream>>>(hbuf, cnt);
  }
  bilstm_scan_kernel<<<2 * NBLK, 256, 0, stream>>>(xb, wcat, bcat, hbuf, cnt, out);
  (void)in_sizes; (void)n_in; (void)out_size; (void)ws_size;
}
